// Seq2SeqModel_29343216566883
// MI455X (gfx1250) — matrix operand order repaired and checked by reading, not compiled, not run on hardware
//
#include <hip/hip_runtime.h>
#include <cstdint>
#include <cstddef>

// Problem dims (fixed by the reference)
#define B_  32
#define S_  128
#define T_  64
#define H_  512
#define E_  512
#define G_  (4*H_)      // 2048 gates
#define VT_ 32000

typedef __attribute__((ext_vector_type(16))) __bf16 v16bf;
typedef __attribute__((ext_vector_type(8)))  float  v8f;

union BfTile { uint4 q[2]; v16bf v; };

__device__ __forceinline__ unsigned short f2bf(float f) {
  unsigned u = __float_as_uint(f);
  u += 0x7FFFu + ((u >> 16) & 1u);          // round-to-nearest-even
  return (unsigned short)(u >> 16);
}
__device__ __forceinline__ float sigf(float x) { return 1.0f / (1.0f + __expf(-x)); }

// ---------------------------------------------------------------------------
// Generic bf16 WMMA GEMM:  C[M,N] (+)= A[M,K] * B[N,K]^T  (+ bias[N])
// A,B row-major bf16 (K contiguous). 4 waves/block; each wave computes a
// 2x2 block of 16x16 tiles (32x32 output) so every A/B fetch feeds two
// v_wmma_f32_16x16x32_bf16 ops (load:WMMA = 2:1, halves weight re-fetch).
// Requires M % 32 == 0 and N % 32 == 0 (true for all call sites).
// ---------------------------------------------------------------------------
__global__ __launch_bounds__(128)
void gemm_bf16_wmma(const unsigned short* __restrict__ A, int lda,
                    const unsigned short* __restrict__ Bw, int ldb,
                    float* __restrict__ C, long long ldc,
                    const float* __restrict__ bias,
                    int M, int N, int K, int accum)
{
  const int lane = threadIdx.x & 31;
  const int wave = threadIdx.x >> 5;
  const int tn = blockIdx.x * 4 + wave;     // 32-wide N block
  const int tm = blockIdx.y;                // 32-wide M block
  if (tn * 32 >= N) return;

  const int half = lane >> 4;               // 0: lanes 0-15, 1: lanes 16-31
  const int mrow0 = tm * 32 + (lane & 15);  // A rows for the two M tiles
  const int mrow1 = mrow0 + 16;
  const int nrow0 = tn * 32 + (lane & 15);  // B (weight) rows for the two N tiles
  const int nrow1 = nrow0 + 16;
  const int akh  = half * 8;                // A: K split 0..7 / 8..15 (+16 block)
  const int bkh  = half * 8;                // B: K split 0..7 / 8..15 (+16 block)

  v8f acc00 = {}, acc01 = {}, acc10 = {}, acc11 = {};
  for (int kk = 0; kk < K; kk += 32) {
    BfTile a0, a1, b0, b1;
    const unsigned short* ap0 = A + (size_t)mrow0 * lda + kk + akh;
    const unsigned short* ap1 = A + (size_t)mrow1 * lda + kk + akh;
    a0.q[0] = *(const uint4*)(ap0);         a0.q[1] = *(const uint4*)(ap0 + 16);
    a1.q[0] = *(const uint4*)(ap1);         a1.q[1] = *(const uint4*)(ap1 + 16);
    const unsigned short* bp0 = Bw + (size_t)nrow0 * ldb + kk + bkh;
    const unsigned short* bp1 = Bw + (size_t)nrow1 * ldb + kk + bkh;
    b0.q[0] = *(const uint4*)(bp0);         b0.q[1] = *(const uint4*)(bp0 + 16);
    b1.q[0] = *(const uint4*)(bp1);         b1.q[1] = *(const uint4*)(bp1 + 16);
    __builtin_prefetch(bp0 + 32, 0, 1);     // next K-chunk of the L2-resident weights
    __builtin_prefetch(bp1 + 32, 0, 1);
    acc00 = __builtin_amdgcn_wmma_f32_16x16x32_bf16(false, a0.v, false, b0.v, (short)0, acc00, false, false);
    acc01 = __builtin_amdgcn_wmma_f32_16x16x32_bf16(false, a0.v, false, b1.v, (short)0, acc01, false, false);
    acc10 = __builtin_amdgcn_wmma_f32_16x16x32_bf16(false, a1.v, false, b0.v, (short)0, acc10, false, false);
    acc11 = __builtin_amdgcn_wmma_f32_16x16x32_bf16(false, a1.v, false, b1.v, (short)0, acc11, false, false);
  }

  const int   ncol0 = tn * 32 + (lane & 15);
  const int   ncol1 = ncol0 + 16;
  const float bv0   = bias ? bias[ncol0] : 0.0f;
  const float bv1   = bias ? bias[ncol1] : 0.0f;
  const int   mb0   = tm * 32 + half * 8;       // C rows for M tile 0
  const int   mb1   = mb0 + 16;                 // C rows for M tile 1
#pragma unroll
  for (int v = 0; v < 8; ++v) {
    size_t r0 = (size_t)(mb0 + v) * (size_t)ldc;
    size_t r1 = (size_t)(mb1 + v) * (size_t)ldc;
    float x00 = acc00[v] + bv0, x01 = acc01[v] + bv1;
    float x10 = acc10[v] + bv0, x11 = acc11[v] + bv1;
    if (accum) {
      x00 += C[r0 + ncol0]; x01 += C[r0 + ncol1];
      x10 += C[r1 + ncol0]; x11 += C[r1 + ncol1];
    }
    C[r0 + ncol0] = x00; C[r0 + ncol1] = x01;
    C[r1 + ncol0] = x10; C[r1 + ncol1] = x11;
  }
}

// ---------------------------------------------------------------------------
// Utility / elementwise kernels
// ---------------------------------------------------------------------------
__global__ void zero_u32_kernel(unsigned int* p, size_t n) {
  size_t i = (size_t)blockIdx.x * blockDim.x + threadIdx.x;
  size_t st = (size_t)gridDim.x * blockDim.x;
  for (; i < n; i += st) p[i] = 0u;
}

__global__ void conv_f32_bf16_kernel(unsigned short* dst, const float* src, size_t n) {
  size_t i = (size_t)blockIdx.x * blockDim.x + threadIdx.x;
  size_t st = (size_t)gridDim.x * blockDim.x;
  for (; i < n; i += st) dst[i] = f2bf(src[i]);
}

// enc_x[s,b,:] = bf16(enc_emb[src[b,s],:])
__global__ void embed_enc_kernel(const int* __restrict__ src,
                                 const float* __restrict__ emb,
                                 unsigned short* __restrict__ encX)
{
  int s = blockIdx.x / B_, b = blockIdx.x % B_;
  int tok = src[b * S_ + s];
  const float* row = emb + (size_t)tok * E_;
  unsigned short* dst = encX + (size_t)blockIdx.x * E_;
  for (int e = threadIdx.x; e < E_; e += blockDim.x) dst[e] = f2bf(row[e]);
}

// LSTM pointwise: gates[B,4H] (i,f,g,o) -> update c, h (fp32 + bf16 + optional y copies)
__global__ void lstm_update_kernel(const float* __restrict__ gates,
                                   float* __restrict__ c,
                                   float* __restrict__ hf,
                                   unsigned short* __restrict__ hbf,
                                   float* __restrict__ yF, long long ysF,
                                   unsigned short* __restrict__ yBf, long long ysBf)
{
  int b = blockIdx.x;
  const float* g = gates + (size_t)b * G_;
  for (int j = threadIdx.x; j < H_; j += blockDim.x) {
    float iv = sigf(g[j]);
    float fv = sigf(g[H_ + j]);
    float gv = tanhf(g[2 * H_ + j]);
    float ov = sigf(g[3 * H_ + j]);
    float cv = fv * c[(size_t)b * H_ + j] + iv * gv;
    c[(size_t)b * H_ + j] = cv;
    float hv = ov * tanhf(cv);
    hf[(size_t)b * H_ + j] = hv;
    unsigned short hb = f2bf(hv);
    hbf[(size_t)b * H_ + j] = hb;
    if (yF)  yF [(size_t)b * ysF  + j] = hv;
    if (yBf) yBf[(size_t)b * ysBf + j] = hb;
  }
}

// scores[b,s] = sum_h v[h] * tanh(qproj[b,h] + enc_proj[b,s,h])
__global__ void attn_score_kernel(const float* __restrict__ qproj,
                                  const float* __restrict__ encProj,
                                  const float* __restrict__ av,
                                  float* __restrict__ scores)
{
  __shared__ float red[128];
  int bs = blockIdx.x;            // b*S + s
  int b  = bs / S_;
  float p = 0.f;
  const float* ep = encProj + (size_t)bs * H_;
  const float* qp = qproj + (size_t)b * H_;
  for (int h = threadIdx.x; h < H_; h += 128)
    p += av[h] * tanhf(qp[h] + ep[h]);
  red[threadIdx.x] = p;
  __syncthreads();
  for (int off = 64; off > 0; off >>= 1) {
    if (threadIdx.x < off) red[threadIdx.x] += red[threadIdx.x + off];
    __syncthreads();
  }
  if (threadIdx.x == 0) scores[bs] = red[0];
}

// softmax over S, write attention weights to output, compute ctx = w @ enc_out
__global__ void softmax_ctx_kernel(const float* __restrict__ scores,
                                   const float* __restrict__ encOutF,
                                   float* __restrict__ wsOut, int t,
                                   float* __restrict__ ctxF,
                                   unsigned short* __restrict__ ctxBf)
{
  __shared__ float w[S_];
  __shared__ float red[S_];
  int b = blockIdx.x, s = threadIdx.x;     // blockDim == S_ == 128
  float sc = scores[b * S_ + s];
  red[s] = sc; __syncthreads();
  for (int off = 64; off > 0; off >>= 1) {
    if (s < off) red[s] = fmaxf(red[s], red[s + off]);
    __syncthreads();
  }
  float mx = red[0]; __syncthreads();
  float e = __expf(sc - mx);
  red[s] = e; __syncthreads();
  for (int off = 64; off > 0; off >>= 1) {
    if (s < off) red[s] += red[s + off];
    __syncthreads();
  }
  float wv = e / red[0];
  w[s] = wv;
  wsOut[((size_t)b * T_ + t) * S_ + s] = wv;
  __syncthreads();
  for (int h = threadIdx.x; h < H_; h += 128) {
    float acc = 0.f;
    const float* eo = encOutF + (size_t)b * S_ * H_ + h;
    for (int s2 = 0; s2 < S_; ++s2) acc += w[s2] * eo[(size_t)s2 * H_];
    ctxF[(size_t)b * H_ + h] = acc;
    ctxBf[(size_t)b * H_ + h] = f2bf(acc);
  }
}

// x0[b] = [ bf16(dec_emb[tok]) , ctx_bf16[b] ] ; tok = (t==0 ? 1 : tgt[b,t-1])
__global__ void dec_embed_kernel(const int* __restrict__ tgt,
                                 const float* __restrict__ emb,
                                 const unsigned short* __restrict__ ctxBf,
                                 unsigned short* __restrict__ x0, int t)
{
  int b = blockIdx.x;
  int tok = (t == 0) ? 1 : tgt[b * T_ + t - 1];
  const float* row = emb + (size_t)tok * E_;
  unsigned short* dst = x0 + (size_t)b * (E_ + H_);
  for (int i = threadIdx.x; i < E_; i += blockDim.x) dst[i] = f2bf(row[i]);
  for (int i = threadIdx.x; i < H_; i += blockDim.x) dst[E_ + i] = ctxBf[(size_t)b * H_ + i];
}

// hc[b] = [ h1_bf16[b] , ctx_bf16[b] ]
__global__ void hc_concat_kernel(const unsigned short* __restrict__ h1bf,
                                 const unsigned short* __restrict__ ctxBf,
                                 unsigned short* __restrict__ hc)
{
  int b = blockIdx.x;
  unsigned short* dst = hc + (size_t)b * (2 * H_);
  for (int i = threadIdx.x; i < H_; i += blockDim.x) {
    dst[i]      = h1bf [(size_t)b * H_ + i];
    dst[H_ + i] = ctxBf[(size_t)b * H_ + i];
  }
}

// ---------------------------------------------------------------------------
// Host orchestration
// ---------------------------------------------------------------------------
extern "C" void kernel_launch(void* const* d_in, const int* in_sizes, int n_in,
                              void* d_out, int out_size, void* d_ws, size_t ws_size,
                              hipStream_t stream)
{
  (void)in_sizes; (void)n_in; (void)out_size; (void)ws_size;

  const int*   src      = (const int*)  d_in[0];
  const int*   tgt      = (const int*)  d_in[1];
  const float* enc_emb  = (const float*)d_in[2];
  const float* enc_Wih0 = (const float*)d_in[3];
  const float* enc_Whh0 = (const float*)d_in[4];
  const float* enc_b0   = (const float*)d_in[5];
  const float* enc_Wih1 = (const float*)d_in[6];
  const float* enc_Whh1 = (const float*)d_in[7];
  const float* enc_b1   = (const float*)d_in[8];
  const float* dec_emb  = (const float*)d_in[9];
  const float* attn_W   = (const float*)d_in[10];
  const float* attn_v   = (const float*)d_in[11];
  const float* dec_Wih0 = (const float*)d_in[12];
  const float* dec_Whh0 = (const float*)d_in[13];
  const float* dec_b0   = (const float*)d_in[14];
  const float* dec_Wih1 = (const float*)d_in[15];
  const float* dec_Whh1 = (const float*)d_in[16];
  const float* dec_b1   = (const float*)d_in[17];
  const float* out_W    = (const float*)d_in[18];
  const float* out_b    = (const float*)d_in[19];

  float* logits = (float*)d_out;                          // [B,T,VT]
  float* wsOut  = logits + (size_t)B_ * T_ * VT_;         // [B,T,S]

  // ---- workspace bump allocator (256B aligned) ----
  char* base = (char*)d_ws;
  size_t off = 0;
  auto alloc = [&](size_t bytes) -> char* {
    off = (off + 255) & ~(size_t)255;
    char* p = base + off;
    off += bytes;
    return p;
  };
  auto allocU16 = [&](size_t n) { return (unsigned short*)alloc(n * 2); };
  auto allocF32 = [&](size_t n) { return (float*)alloc(n * 4); };

  unsigned short* encX     = allocU16((size_t)S_ * B_ * E_);
  unsigned short* y0bf     = allocU16((size_t)S_ * B_ * H_);
  unsigned short* encOutBf = allocU16((size_t)B_ * S_ * H_);
  unsigned short* wEncWih0 = allocU16((size_t)G_ * E_);
  unsigned short* wEncWhh0 = allocU16((size_t)G_ * H_);
  unsigned short* wEncWih1 = allocU16((size_t)G_ * H_);
  unsigned short* wEncWhh1 = allocU16((size_t)G_ * H_);
  unsigned short* wDecWih0 = allocU16((size_t)G_ * (E_ + H_));
  unsigned short* wDecWhh0 = allocU16((size_t)G_ * H_);
  unsigned short* wDecWih1 = allocU16((size_t)G_ * H_);
  unsigned short* wDecWhh1 = allocU16((size_t)G_ * H_);
  unsigned short* wAttn    = allocU16((size_t)H_ * 2 * H_);
  unsigned short* wOut     = allocU16((size_t)VT_ * 2 * H_);
  unsigned short* h0bf     = allocU16((size_t)B_ * H_);
  unsigned short* h1bf     = allocU16((size_t)B_ * H_);
  unsigned short* ctxBf    = allocU16((size_t)B_ * H_);
  unsigned short* x0       = allocU16((size_t)B_ * (E_ + H_));
  unsigned short* hc       = allocU16((size_t)B_ * 2 * H_);

  float* encOutF  = allocF32((size_t)B_ * S_ * H_);
  float* encProjF = allocF32((size_t)B_ * S_ * H_);
  float* gates    = allocF32((size_t)B_ * G_);
  float* h0f      = allocF32((size_t)B_ * H_);
  float* c0       = allocF32((size_t)B_ * H_);
  float* h1f      = allocF32((size_t)B_ * H_);
  float* c1       = allocF32((size_t)B_ * H_);
  float* qprojF   = allocF32((size_t)B_ * H_);
  float* scores   = allocF32((size_t)B_ * S_);
  float* ctxF     = allocF32((size_t)B_ * H_);

  auto gemm = [&](const unsigned short* A, int lda, const unsigned short* Bw, int ldb,
                  float* C, long long ldc, const float* bias,
                  int M, int N, int K, int accum) {
    dim3 grid((N / 32 + 3) / 4, M / 32);
    gemm_bf16_wmma<<<grid, 128, 0, stream>>>(A, lda, Bw, ldb, C, ldc, bias, M, N, K, accum);
  };
  auto conv = [&](unsigned short* dst, const float* s, size_t n) {
    int blocks = (int)((n + 256 * 8 - 1) / (256 * 8));
    if (blocks > 4096) blocks = 4096;
    conv_f32_bf16_kernel<<<blocks, 256, 0, stream>>>(dst, s, n);
  };
  auto zero = [&](void* p, size_t bytes) {
    size_t n = bytes / 4;
    int blocks = (int)((n + 255) / 256);
    if (blocks > 1024) blocks = 1024;
    zero_u32_kernel<<<blocks, 256, 0, stream>>>((unsigned int*)p, n);
  };

  // ---- one-time per call: bf16 weight copies (out_W stays L2-resident) ----
  conv(wEncWih0, enc_Wih0, (size_t)G_ * E_);
  conv(wEncWhh0, enc_Whh0, (size_t)G_ * H_);
  conv(wEncWih1, enc_Wih1, (size_t)G_ * H_);
  conv(wEncWhh1, enc_Whh1, (size_t)G_ * H_);
  conv(wDecWih0, dec_Wih0, (size_t)G_ * (E_ + H_));
  conv(wDecWhh0, dec_Whh0, (size_t)G_ * H_);
  conv(wDecWih1, dec_Wih1, (size_t)G_ * H_);
  conv(wDecWhh1, dec_Whh1, (size_t)G_ * H_);
  conv(wAttn,    attn_W,   (size_t)H_ * 2 * H_);
  conv(wOut,     out_W,    (size_t)VT_ * 2 * H_);

  embed_enc_kernel<<<S_ * B_, 128, 0, stream>>>(src, enc_emb, encX);

  zero(h0f, (size_t)B_ * H_ * 4);  zero(c0, (size_t)B_ * H_ * 4);
  zero(h1f, (size_t)B_ * H_ * 4);  zero(c1, (size_t)B_ * H_ * 4);
  zero(h0bf, (size_t)B_ * H_ * 2); zero(h1bf, (size_t)B_ * H_ * 2);

  // ---- encoder: 2-layer LSTM over S steps ----
  for (int s = 0; s < S_; ++s) {
    // layer 0: gates = x_s @ Wih0^T + b0 ; += h0 @ Whh0^T
    gemm(encX + (size_t)s * B_ * E_, E_, wEncWih0, E_, gates, G_, enc_b0, B_, G_, E_, 0);
    gemm(h0bf, H_, wEncWhh0, H_, gates, G_, nullptr, B_, G_, H_, 1);
    lstm_update_kernel<<<B_, 128, 0, stream>>>(gates, c0, h0f, h0bf,
        nullptr, 0, y0bf + (size_t)s * B_ * H_, H_);
    // layer 1
    gemm(y0bf + (size_t)s * B_ * H_, H_, wEncWih1, H_, gates, G_, enc_b1, B_, G_, H_, 0);
    gemm(h1bf, H_, wEncWhh1, H_, gates, G_, nullptr, B_, G_, H_, 1);
    lstm_update_kernel<<<B_, 128, 0, stream>>>(gates, c1, h1f, h1bf,
        encOutF + (size_t)s * H_, (long long)S_ * H_,
        encOutBf + (size_t)s * H_, (long long)S_ * H_);
  }

  // ---- enc_proj = enc_out @ Wa_e^T  (Wa_e = attn_W[:, H:], ldb = 2H) ----
  gemm(encOutBf, H_, wAttn + H_, 2 * H_, encProjF, H_, nullptr, B_ * S_, H_, H_, 0);

  // ---- decoder: T steps with concat attention + 2-layer LSTM + vocab proj ----
  for (int t = 0; t < T_; ++t) {
    // attention query from carry h1
    gemm(h1bf, H_, wAttn, 2 * H_, qprojF, H_, nullptr, B_, H_, H_, 0);
    attn_score_kernel<<<B_ * S_, 128, 0, stream>>>(qprojF, encProjF, attn_v, scores);
    softmax_ctx_kernel<<<B_, 128, 0, stream>>>(scores, encOutF, wsOut, t, ctxF, ctxBf);
    dec_embed_kernel<<<B_, 256, 0, stream>>>(tgt, dec_emb, ctxBf, x0, t);

    // layer 0 (K = E+H = 1024)
    gemm(x0, E_ + H_, wDecWih0, E_ + H_, gates, G_, dec_b0, B_, G_, E_ + H_, 0);
    gemm(h0bf, H_, wDecWhh0, H_, gates, G_, nullptr, B_, G_, H_, 1);
    lstm_update_kernel<<<B_, 128, 0, stream>>>(gates, c0, h0f, h0bf, nullptr, 0, nullptr, 0);
    // layer 1
    gemm(h0bf, H_, wDecWih1, H_, gates, G_, dec_b1, B_, G_, H_, 0);
    gemm(h1bf, H_, wDecWhh1, H_, gates, G_, nullptr, B_, G_, H_, 1);
    lstm_update_kernel<<<B_, 128, 0, stream>>>(gates, c1, h1f, h1bf, nullptr, 0, nullptr, 0);

    // logits = [h1; ctx] @ out_W^T + out_b  -> straight into d_out[:, t, :]
    hc_concat_kernel<<<B_, 256, 0, stream>>>(h1bf, ctxBf, hc);
    gemm(hc, 2 * H_, wOut, 2 * H_, logits + (size_t)t * VT_, (long long)T_ * VT_,
         out_b, B_, VT_, 2 * H_, 0);
  }
}
